// MemoryEfficientMambaBlock_88467736363285
// MI455X (gfx1250) — matrix operand order repaired and checked by reading, not compiled, not run on hardware
//
#include <hip/hip_runtime.h>
#include <hip/hip_bf16.h>
#include <math.h>

// ---------------------------------------------------------------------------
// MemoryEfficientMambaBlock fused kernel for gfx1250 (MI455X, wave32, WMMA)
// ---------------------------------------------------------------------------

typedef _Float16 h4   __attribute__((ext_vector_type(4)));
typedef _Float16 h8   __attribute__((ext_vector_type(8)));
typedef _Float16 v16h __attribute__((ext_vector_type(16)));
typedef float    v8f  __attribute__((ext_vector_type(8)));

#define D_MODEL    1024
#define D_INNER    2048
#define D_STATE    8
#define TOKENS     50176          // 2*128*196
#define TOK_BLK    64             // tokens per block
#define A_STRIDE   1032           // f16 elems per LDS row (1024 + 8 pad, bank-friendly)
#define SCR_STRIDE 68             // f32 per scratch row (64 + 4 pad)
#define N_WAVES    8

#define LDS_A_BYTES    (TOK_BLK * A_STRIDE * 2)                 // 132096
#define LDS_SCR_BYTES  (N_WAVES * 16 * SCR_STRIDE * 4)          // 34816
#define LDS_ST_BYTES   (TOK_BLK * D_STATE * 4)                  // 2048
#define LDS_BYTES      (LDS_A_BYTES + LDS_SCR_BYTES + LDS_ST_BYTES)

union AF { v16h v; h8 h[2]; };

__device__ __forceinline__ float wave_allreduce_add(float v) {
    v += __shfl_xor(v, 16);
    v += __shfl_xor(v, 8);
    v += __shfl_xor(v, 4);
    v += __shfl_xor(v, 2);
    v += __shfl_xor(v, 1);
    return v;
}

__device__ __forceinline__ float silu_f(float v) {
    return v / (1.0f + __expf(-v));
}

// --- tiny weight cast: W_proj f32 -> f16 (stays resident in L2, 4 MB) ------
__global__ void wproj_cast_f16(const float* __restrict__ w,
                               _Float16* __restrict__ o, int n4) {
    int i = blockIdx.x * blockDim.x + threadIdx.x;
    if (i < n4) {
        float4 v = ((const float4*)w)[i];
        h4 hv;
        hv.x = (_Float16)v.x; hv.y = (_Float16)v.y;
        hv.z = (_Float16)v.z; hv.w = (_Float16)v.w;
        ((h4*)o)[i] = hv;
    }
}

// --- fused LN + GEMM(f16 WMMA) + silu + state proj + silu + out proj -------
extern "C" __global__ __launch_bounds__(256, 1)
void mamba_fused(const float* __restrict__ x,
                 const _Float16* __restrict__ Wp16,     // [2048][1024] f16
                 const float* __restrict__ b_proj,      // [2048]
                 const float* __restrict__ W_state,     // [8][2048]
                 const float* __restrict__ b_state,     // [8]
                 const float* __restrict__ W_out,       // [1024][8]
                 const float* __restrict__ b_out,       // [1024]
                 const float* __restrict__ init_state,  // [8]
                 const float* __restrict__ gamma,       // [1024]
                 const float* __restrict__ beta,        // [1024]
                 float* __restrict__ out)
{
    extern __shared__ __align__(16) char smem_raw[];
    _Float16* aTile   = (_Float16*)smem_raw;                       // [64][1032] f16
    float*    scratch = (float*)(smem_raw + LDS_A_BYTES);          // [8 waves][16][68]
    float*    stateAcc= (float*)(smem_raw + LDS_A_BYTES + LDS_SCR_BYTES); // [64][8]

    const int tid  = threadIdx.x;
    const int wave = tid >> 5;
    const int lane = tid & 31;
    const int hi   = lane >> 4;          // half-wave select
    const int lo16 = lane & 15;
    const size_t tok0 = (size_t)blockIdx.x * TOK_BLK;

    // zero the cross-wave state accumulator (512 floats)
    stateAcc[tid]       = 0.0f;
    stateAcc[tid + 256] = 0.0f;

    // ---------------- Phase 1: LayerNorm -> f16 A-tile in LDS --------------
    // wave w normalizes tokens w*8 .. w*8+7 (32 lanes x 8 float4 each)
    for (int i = 0; i < 8; ++i) {
        const int t = wave * 8 + i;
        const float4* xr = (const float4*)(x + (tok0 + t) * D_MODEL);
        float4 xv[8];
        float s = 0.0f, sq = 0.0f;
        #pragma unroll
        for (int j = 0; j < 8; ++j) {
            xv[j] = xr[j * 32 + lane];
            s  += xv[j].x + xv[j].y + xv[j].z + xv[j].w;
            sq += xv[j].x * xv[j].x + xv[j].y * xv[j].y
                + xv[j].z * xv[j].z + xv[j].w * xv[j].w;
        }
        s  = wave_allreduce_add(s);
        sq = wave_allreduce_add(sq);
        const float mean = s * (1.0f / (float)D_MODEL);
        const float var  = sq * (1.0f / (float)D_MODEL) - mean * mean;
        const float rstd = rsqrtf(var + 1e-5f);
        #pragma unroll
        for (int j = 0; j < 8; ++j) {
            const int e4 = j * 32 + lane;          // float4 index within the row
            float4 g = ((const float4*)gamma)[e4];
            float4 b = ((const float4*)beta)[e4];
            h4 hv;
            hv.x = (_Float16)((xv[j].x - mean) * rstd * g.x + b.x);
            hv.y = (_Float16)((xv[j].y - mean) * rstd * g.y + b.y);
            hv.z = (_Float16)((xv[j].z - mean) * rstd * g.z + b.z);
            hv.w = (_Float16)((xv[j].w - mean) * rstd * g.w + b.w);
            *(h4*)(aTile + (size_t)t * A_STRIDE + (size_t)e4 * 4) = hv;
        }
    }
    __syncthreads();

    // ---------------- Phase 2: WMMA GEMM + fused state reduction -----------
    // wave nt owns inner columns {n0 + nt*16 + 0..15} for each 128-wide chunk.
    const int nt = wave;
    float sacc0[D_STATE], sacc1[D_STATE];
    #pragma unroll
    for (int s = 0; s < D_STATE; ++s) { sacc0[s] = 0.0f; sacc1[s] = 0.0f; }
    const int tA = 2 * lane;         // each lane owns tokens tA, tB across all chunks
    const int tB = 2 * lane + 1;
    float* wscr = scratch + wave * 16 * SCR_STRIDE;

    for (int chunk = 0; chunk < D_INNER / 128; ++chunk) {
        const int n0   = chunk * 128;
        const int nIdx = n0 + nt * 16 + lo16;       // inner column owned by this lane
        const _Float16* bRow = Wp16 + (size_t)nIdx * D_MODEL;

        if (chunk + 1 < D_INNER / 128)               // warm L2->L0 path for next B slice
            __builtin_prefetch(Wp16 + (size_t)(nIdx + 128) * D_MODEL, 0, 0);

        v8f c[4];
        #pragma unroll
        for (int mt = 0; mt < 4; ++mt) c[mt] = (v8f){0,0,0,0,0,0,0,0};

        for (int k0 = 0; k0 < D_MODEL; k0 += 32) {
            // B fragment (32x16 f16): half-wave holds contiguous 16 K values
            AF b;
            const _Float16* bp = bRow + k0 + hi * 16;
            b.h[0] = *(const h8*)(bp);
            b.h[1] = *(const h8*)(bp + 8);
            #pragma unroll
            for (int mt = 0; mt < 4; ++mt) {
                // A fragment (16x32 f16): half-wave holds contiguous 16 K values
                AF a;
                const _Float16* ap = aTile + (size_t)(mt * 16 + lo16) * A_STRIDE
                                   + k0 + hi * 16;
                a.h[0] = *(const h8*)(ap);
                a.h[1] = *(const h8*)(ap + 8);
                c[mt] = __builtin_amdgcn_wmma_f32_16x16x32_f16(
                    false, a.v, false, b.v, (short)0, c[mt], false, false);
            }
        }

        // bias + SiLU, transpose through per-wave LDS scratch [n_local][token]
        const float bp = b_proj[nIdx];
        #pragma unroll
        for (int mt = 0; mt < 4; ++mt) {
            #pragma unroll
            for (int r = 0; r < 8; ++r) {
                const int tok = mt * 16 + r + 8 * hi;     // C layout: M = r + 8*hi
                wscr[lo16 * SCR_STRIDE + tok] = silu_f(c[mt][r] + bp);
            }
        }
        // DS ops are in-order within a wave: reads below see the stores above.

        // fold this wave's 16 inner columns into per-lane state partials
        #pragma unroll
        for (int n = 0; n < 16; ++n) {
            const float va = wscr[n * SCR_STRIDE + tA];
            const float vb = wscr[n * SCR_STRIDE + tB];
            const int ng = n0 + nt * 16 + n;
            #pragma unroll
            for (int s = 0; s < D_STATE; ++s) {
                const float w = W_state[s * D_INNER + ng];
                sacc0[s] = fmaf(va, w, sacc0[s]);
                sacc1[s] = fmaf(vb, w, sacc1[s]);
            }
        }
    }

    // merge the 8 waves' partial states (once per block)
    #pragma unroll
    for (int s = 0; s < D_STATE; ++s) {
        atomicAdd(&stateAcc[tA * D_STATE + s], sacc0[s]);
        atomicAdd(&stateAcc[tB * D_STATE + s], sacc1[s]);
    }
    __syncthreads();

    // ---------------- Phase 3: silu(state) -> out projection + residual ----
    const int t    = tid >> 2;               // token 0..63 (4 threads per token)
    const int dseg = (tid & 3) * 256;        // 256 output dims per thread
    float cs[D_STATE];
    #pragma unroll
    for (int s = 0; s < D_STATE; ++s)
        cs[s] = silu_f(stateAcc[t * D_STATE + s] + b_state[s] + init_state[s]);

    const size_t tg = tok0 + t;
    const float4* xr = (const float4*)(x + tg * D_MODEL);
    float4* orow = (float4*)(out + tg * D_MODEL);
    for (int d4 = dseg / 4; d4 < dseg / 4 + 64; ++d4) {
        const float4 xv = xr[d4];
        const float4 bo = ((const float4*)b_out)[d4];
        float acc[4];
        #pragma unroll
        for (int j = 0; j < 4; ++j) {
            const float* wr = W_out + (size_t)(d4 * 4 + j) * D_STATE;
            float a = 0.0f;
            #pragma unroll
            for (int s = 0; s < D_STATE; ++s) a = fmaf(cs[s], wr[s], a);
            acc[j] = a;
        }
        float4 r;
        r.x = xv.x + bo.x + acc[0];
        r.y = xv.y + bo.y + acc[1];
        r.z = xv.z + bo.z + acc[2];
        r.w = xv.w + bo.w + acc[3];
        orow[d4] = r;
    }
}

// ---------------------------------------------------------------------------
extern "C" void kernel_launch(void* const* d_in, const int* in_sizes, int n_in,
                              void* d_out, int out_size, void* d_ws, size_t ws_size,
                              hipStream_t stream) {
    const float* x        = (const float*)d_in[0];
    const float* W_proj   = (const float*)d_in[1];
    const float* b_proj   = (const float*)d_in[2];
    const float* W_state  = (const float*)d_in[3];
    const float* b_state  = (const float*)d_in[4];
    const float* W_out    = (const float*)d_in[5];
    const float* b_out    = (const float*)d_in[6];
    const float* init_st  = (const float*)d_in[7];
    const float* gamma    = (const float*)d_in[8];
    const float* beta     = (const float*)d_in[9];
    float* out            = (float*)d_out;

    _Float16* Wp16 = (_Float16*)d_ws;                 // 4 MB f16 weight cache

    const int n4 = (D_INNER * D_MODEL) / 4;
    wproj_cast_f16<<<(n4 + 255) / 256, 256, 0, stream>>>(W_proj, Wp16, n4);

    // 169 KB dynamic LDS per block — CDNA5 WGP offers 320 KB.
    (void)hipFuncSetAttribute((const void*)mamba_fused,
                              hipFuncAttributeMaxDynamicSharedMemorySize,
                              (int)LDS_BYTES);

    const int nblocks = TOKENS / TOK_BLK;             // 784
    mamba_fused<<<nblocks, 256, LDS_BYTES, stream>>>(
        x, Wp16, b_proj, W_state, b_state, W_out, b_out, init_st,
        gamma, beta, out);
}
